// MoE_9079560863781
// MI455X (gfx1250) — compile-verified, matrix operand order checked by reading, not run on hardware
//
#include <hip/hip_runtime.h>

#define NTOK 4096
#define DDIM 1024
#define HDIM 2048
#define NEXP 8

typedef __attribute__((ext_vector_type(16))) __bf16   v16bf;
typedef __attribute__((ext_vector_type(8)))  float    v8f;
typedef __attribute__((ext_vector_type(4)))  unsigned int v4u;
typedef __attribute__((ext_vector_type(4)))  float    v4f;

union FragBF { v16bf v; v4u q[2]; };

// ---------------------------------------------------------------- zero
__global__ void zero_kernel(float* __restrict__ y,
                            int* __restrict__ cnt, float* __restrict__ imp) {
    int i = blockIdx.x * blockDim.x + threadIdx.x;
    if (i < NEXP) { cnt[i] = 0; imp[i] = 0.0f; }
    if (i == 0) y[NTOK * DDIM] = 0.0f;               // loss slot
    v4f z = {0.f, 0.f, 0.f, 0.f};
    for (int k = i; k < NTOK * DDIM / 4; k += gridDim.x * blockDim.x)
        *(v4f*)(y + (size_t)k * 4) = z;
}

// ---------------------------------------------------------------- x -> bf16
__global__ __launch_bounds__(256) void conv_x_kernel(const float* __restrict__ x,
                                                     __bf16* __restrict__ xb) {
    int i = (blockIdx.x * blockDim.x + threadIdx.x) * 8;
#pragma unroll
    for (int j = 0; j < 8; j++) xb[i + j] = (__bf16)x[i + j];
}

// ------------------------------------------------- weight transpose + bf16
// src: [E][R][C] f32 (row-major over C)  ->  dst: [E][C][R] bf16
__global__ __launch_bounds__(256) void conv_tr_kernel(const float* __restrict__ src,
                                                      __bf16* __restrict__ dst,
                                                      int R, int C) {
    const int e  = blockIdx.z;
    const int r0 = blockIdx.x * 64;
    const int c0 = blockIdx.y * 64;
    __shared__ __bf16 tile[64][80];
    const int tid = threadIdx.x;
    {
        int r  = tid >> 2;
        int cc = (tid & 3) * 16;
        const float* s = src + ((size_t)e * R + r0 + r) * C + c0 + cc;
#pragma unroll
        for (int j = 0; j < 16; j++) tile[r][cc + j] = (__bf16)s[j];
    }
    __syncthreads();
    {
        int c  = tid >> 2;
        int rc = (tid & 3) * 16;
        __bf16* d = dst + ((size_t)e * C + c0 + c) * R + r0 + rc;
#pragma unroll
        for (int j = 0; j < 16; j++) d[j] = tile[rc + j][c];
    }
}

// ---------------------------------------------------------------- gating
__global__ __launch_bounds__(256) void gate_kernel(
        const float* __restrict__ x, const float* __restrict__ wg_g,
        int* __restrict__ cnt, float* __restrict__ imp,
        int* __restrict__ list_tok, float* __restrict__ list_gate) {
    __shared__ float wg[DDIM * NEXP];
    const int tid = threadIdx.x;
    for (int i = tid; i < DDIM * NEXP; i += blockDim.x) wg[i] = wg_g[i];
    __syncthreads();

    const int t = blockIdx.x * blockDim.x + tid;
    float acc[NEXP];
#pragma unroll
    for (int e = 0; e < NEXP; e++) acc[e] = 0.0f;
    const float* xr = x + (size_t)t * DDIM;
    for (int d = 0; d < DDIM; d++) {
        float xv = xr[d];
#pragma unroll
        for (int e = 0; e < NEXP; e++) acc[e] += xv * wg[d * NEXP + e];
    }
    int i0 = 0; float v0 = acc[0];
#pragma unroll
    for (int e = 1; e < NEXP; e++) if (acc[e] > v0) { v0 = acc[e]; i0 = e; }
    int i1 = (i0 == 0) ? 1 : 0; float v1 = acc[i1];
#pragma unroll
    for (int e = 0; e < NEXP; e++)
        if (e != i0 && acc[e] > v1) { v1 = acc[e]; i1 = e; }
    float e1  = __expf(v1 - v0);
    float inv = 1.0f / (1.0f + e1);
    float g0 = inv, g1 = e1 * inv;

    int s0 = atomicAdd(&cnt[i0], 1);
    list_tok [i0 * NTOK + s0] = t;
    list_gate[i0 * NTOK + s0] = g0;
    unsafeAtomicAdd(&imp[i0], g0);

    int s1 = atomicAdd(&cnt[i1], 1);
    list_tok [i1 * NTOK + s1] = t;
    list_gate[i1 * NTOK + s1] = g1;
    unsafeAtomicAdd(&imp[i1], g1);
}

// ---------------------------------------------------------------- bases + loss
__global__ void base_loss_kernel(const int* __restrict__ cnt,
                                 const float* __restrict__ imp,
                                 int* __restrict__ base,
                                 float* __restrict__ loss_out) {
    if (threadIdx.x == 0 && blockIdx.x == 0) {
        int b = 0; float mi = 0.0f, ml = 0.0f;
        for (int e = 0; e < NEXP; e++) {
            base[e] = b; b += cnt[e];
            mi += imp[e]; ml += (float)cnt[e];
        }
        mi /= NEXP; ml /= NEXP;
        float vi = 0.0f, vl = 0.0f;
        for (int e = 0; e < NEXP; e++) {
            float di = imp[e] - mi;        vi += di * di;
            float dl = (float)cnt[e] - ml; vl += dl * dl;
        }
        vi /= (NEXP - 1); vl /= (NEXP - 1);
        *loss_out = 0.01f * (vi / (mi * mi + 1e-10f) + vl / (ml * ml + 1e-10f));
    }
}

// ---------------------------------------------------------------- FFN stage 1
// h[slot,:] = silu((X@w1+b1)*(X@w3+b3)); w1t/w3t are [E][H][D] bf16 (k-contig)
__global__ __launch_bounds__(256) void ffn1_kernel(
        const __bf16* __restrict__ xb,
        const __bf16* __restrict__ w1t, const float* __restrict__ b1,
        const __bf16* __restrict__ w3t, const float* __restrict__ b3,
        const int* __restrict__ cnt, const int* __restrict__ base,
        const int* __restrict__ list_tok, __bf16* __restrict__ hbuf) {
    const int e     = blockIdx.z;
    const int cnt_e = cnt[e];
    const int m0    = blockIdx.x * 64;
    if (m0 >= cnt_e) return;
    const int n0     = blockIdx.y * 64;
    const int base_e = base[e];

    __shared__ __bf16 As [2][64][40];
    __shared__ __bf16 B1s[2][64][40];
    __shared__ __bf16 B3s[2][64][40];

    const int tid  = threadIdx.x;
    const int lane = tid & 31;
    const int wave = tid >> 5;
    const int mi   = wave & 3;
    const int nc0  = (wave >> 2) * 2;
    const int lm   = lane & 15;
    const int lh   = lane >> 4;

    v8f accU[2] = {};
    v8f accV[2] = {};

    const int tr = tid >> 2;          // tile row 0..63
    const int tk = (tid & 3) * 8;     // k-chunk 0/8/16/24
    // clamp gather row: out-of-range rows produce dead output rows (guarded at store)
    const int arow_i = (m0 + tr < cnt_e) ? (m0 + tr) : (cnt_e - 1);
    const int tokA   = list_tok[e * NTOK + arow_i];
    const __bf16* arow  = xb  + (size_t)tokA * DDIM + tk;
    const __bf16* b1row = w1t + ((size_t)e * HDIM + n0 + tr) * DDIM + tk;
    const __bf16* b3row = w3t + ((size_t)e * HDIM + n0 + tr) * DDIM + tk;

    v4u ra  = *(const v4u*)(arow);
    v4u rb1 = *(const v4u*)(b1row);
    v4u rb3 = *(const v4u*)(b3row);

    int buf = 0;
    for (int k0 = 0; k0 < DDIM; k0 += 32) {
        *(v4u*)&As [buf][tr][tk] = ra;
        *(v4u*)&B1s[buf][tr][tk] = rb1;
        *(v4u*)&B3s[buf][tr][tk] = rb3;
        __syncthreads();

        const int k1 = k0 + 32;
        if (k1 < DDIM) {                       // uniform branch
            ra  = *(const v4u*)(arow  + k1);
            rb1 = *(const v4u*)(b1row + k1);
            rb3 = *(const v4u*)(b3row + k1);
            __builtin_prefetch(b1row + k1 + 32, 0, 3);   // WGP-scope prefetch
            __builtin_prefetch(b3row + k1 + 32, 0, 3);
        }

        FragBF a;
        a.q[0] = *(const v4u*)&As[buf][mi * 16 + lm][lh * 8];
        a.q[1] = *(const v4u*)&As[buf][mi * 16 + lm][16 + lh * 8];
#pragma unroll
        for (int c = 0; c < 2; c++) {
            FragBF b;
            b.q[0] = *(const v4u*)&B1s[buf][(nc0 + c) * 16 + lm][lh * 8];
            b.q[1] = *(const v4u*)&B1s[buf][(nc0 + c) * 16 + lm][16 + lh * 8];
            accU[c] = __builtin_amdgcn_wmma_f32_16x16x32_bf16(
                false, a.v, false, b.v, (short)0, accU[c], false, false);
            b.q[0] = *(const v4u*)&B3s[buf][(nc0 + c) * 16 + lm][lh * 8];
            b.q[1] = *(const v4u*)&B3s[buf][(nc0 + c) * 16 + lm][16 + lh * 8];
            accV[c] = __builtin_amdgcn_wmma_f32_16x16x32_bf16(
                false, a.v, false, b.v, (short)0, accV[c], false, false);
        }
        buf ^= 1;
    }

#pragma unroll
    for (int r = 0; r < 8; r++) {
        int row = m0 + mi * 16 + lh * 8 + r;
        if (row < cnt_e) {
#pragma unroll
            for (int c = 0; c < 2; c++) {
                int col  = n0 + (nc0 + c) * 16 + lm;
                float u  = accU[c][r] + b1[(size_t)e * HDIM + col];
                float vv = accV[c][r] + b3[(size_t)e * HDIM + col];
                float p  = u * vv;
                float hv = p / (1.0f + __expf(-p));      // silu(u*v)
                hbuf[(size_t)(base_e + row) * HDIM + col] = (__bf16)hv;
            }
        }
    }
}

// ---------------------------------------------------------------- FFN stage 2
// y[tok,:] += gate * (h @ w2 + b2); w2t is [E][D][H] bf16 (k-contig)
__global__ __launch_bounds__(256) void ffn2_kernel(
        const __bf16* __restrict__ hbuf,
        const __bf16* __restrict__ w2t, const float* __restrict__ b2,
        const int* __restrict__ cnt, const int* __restrict__ base,
        const int* __restrict__ list_tok, const float* __restrict__ list_gate,
        float* __restrict__ y) {
    const int e     = blockIdx.z;
    const int cnt_e = cnt[e];
    const int m0    = blockIdx.x * 64;
    if (m0 >= cnt_e) return;
    const int d0     = blockIdx.y * 64;
    const int base_e = base[e];

    __shared__ __bf16 As[2][64][40];
    __shared__ __bf16 Bs[2][64][40];

    const int tid  = threadIdx.x;
    const int lane = tid & 31;
    const int wave = tid >> 5;
    const int mi   = wave & 3;
    const int nc0  = (wave >> 2) * 2;
    const int lm   = lane & 15;
    const int lh   = lane >> 4;

    v8f acc[2] = {};

    const int tr = tid >> 2;
    const int tk = (tid & 3) * 8;
    const int arow_i = (m0 + tr < cnt_e) ? (m0 + tr) : (cnt_e - 1);
    const __bf16* hrow = hbuf + (size_t)(base_e + arow_i) * HDIM + tk;
    const __bf16* brow = w2t + ((size_t)e * DDIM + d0 + tr) * HDIM + tk;

    v4u ra = *(const v4u*)(hrow);
    v4u rb = *(const v4u*)(brow);

    int buf = 0;
    for (int k0 = 0; k0 < HDIM; k0 += 32) {
        *(v4u*)&As[buf][tr][tk] = ra;
        *(v4u*)&Bs[buf][tr][tk] = rb;
        __syncthreads();

        const int k1 = k0 + 32;
        if (k1 < HDIM) {                       // uniform branch
            ra = *(const v4u*)(hrow + k1);
            rb = *(const v4u*)(brow + k1);
            __builtin_prefetch(brow + k1 + 32, 0, 3);    // WGP-scope prefetch
        }

        FragBF a;
        a.q[0] = *(const v4u*)&As[buf][mi * 16 + lm][lh * 8];
        a.q[1] = *(const v4u*)&As[buf][mi * 16 + lm][16 + lh * 8];
#pragma unroll
        for (int c = 0; c < 2; c++) {
            FragBF b;
            b.q[0] = *(const v4u*)&Bs[buf][(nc0 + c) * 16 + lm][lh * 8];
            b.q[1] = *(const v4u*)&Bs[buf][(nc0 + c) * 16 + lm][16 + lh * 8];
            acc[c] = __builtin_amdgcn_wmma_f32_16x16x32_bf16(
                false, a.v, false, b.v, (short)0, acc[c], false, false);
        }
        buf ^= 1;
    }

#pragma unroll
    for (int r = 0; r < 8; r++) {
        int row = m0 + mi * 16 + lh * 8 + r;
        if (row < cnt_e) {
            int   tok  = list_tok [e * NTOK + row];
            float gate = list_gate[e * NTOK + row];
#pragma unroll
            for (int c = 0; c < 2; c++) {
                int col   = d0 + (nc0 + c) * 16 + lm;
                float val = gate * (acc[c][r] + b2[(size_t)e * DDIM + col]);
                unsafeAtomicAdd(&y[(size_t)tok * DDIM + col], val);
            }
        }
    }
}

// ---------------------------------------------------------------- launch
extern "C" void kernel_launch(void* const* d_in, const int* in_sizes, int n_in,
                              void* d_out, int out_size, void* d_ws, size_t ws_size,
                              hipStream_t stream) {
    const float* x      = (const float*)d_in[0];
    const float* w_gate = (const float*)d_in[1];
    const float* w1     = (const float*)d_in[2];
    const float* b1     = (const float*)d_in[3];
    const float* w3     = (const float*)d_in[4];
    const float* b3     = (const float*)d_in[5];
    const float* w2     = (const float*)d_in[6];
    const float* b2     = (const float*)d_in[7];
    float* y = (float*)d_out;

    // workspace layout (bytes)
    char* ws = (char*)d_ws;
    int*    cnt       = (int*)ws;                            // 8 ints
    int*    base      = cnt + 8;                             // 8 ints
    float*  imp       = (float*)(base + 8);                  // 8 floats
    int*    list_tok  = (int*)(ws + 256);                    // 128 KB
    float*  list_gate = (float*)(ws + 256 + NEXP * NTOK * 4);// 128 KB
    __bf16* xb   = (__bf16*)(ws + 262400);                   // 8 MB
    __bf16* w1t  = (__bf16*)(ws + 262400 +  8388608);        // 32 MB  [E][H][D]
    __bf16* w3t  = (__bf16*)(ws + 262400 + 41943040);        // 32 MB  [E][H][D]
    __bf16* w2t  = (__bf16*)(ws + 262400 + 75497472);        // 32 MB  [E][D][H]
    __bf16* hbuf = (__bf16*)(ws + 262400 + 109051904);       // 32 MB

    zero_kernel<<<4096, 256, 0, stream>>>(y, cnt, imp);
    conv_x_kernel<<<NTOK * DDIM / 2048, 256, 0, stream>>>(x, xb);
    conv_tr_kernel<<<dim3(DDIM / 64, HDIM / 64, NEXP), 256, 0, stream>>>(w1, w1t, DDIM, HDIM);
    conv_tr_kernel<<<dim3(DDIM / 64, HDIM / 64, NEXP), 256, 0, stream>>>(w3, w3t, DDIM, HDIM);
    conv_tr_kernel<<<dim3(HDIM / 64, DDIM / 64, NEXP), 256, 0, stream>>>(w2, w2t, HDIM, DDIM);
    gate_kernel<<<NTOK / 256, 256, 0, stream>>>(x, w_gate, cnt, imp, list_tok, list_gate);
    base_loss_kernel<<<1, 64, 0, stream>>>(cnt, imp, base, y + (size_t)NTOK * DDIM);
    ffn1_kernel<<<dim3(NTOK / 64, HDIM / 64, NEXP), 256, 0, stream>>>(
        xb, w1t, b1, w3t, b3, cnt, base, list_tok, hbuf);
    ffn2_kernel<<<dim3(NTOK / 64, DDIM / 64, NEXP), 256, 0, stream>>>(
        hbuf, w2t, b2, cnt, base, list_tok, list_gate, y);
}
